// MaskPromptImageNetViT_63213328663265
// MI455X (gfx1250) — matrix operand order repaired and checked by reading, not compiled, not run on hardware
//
#include <hip/hip_runtime.h>
#include <hip/hip_bf16.h>

// ---------------------------------------------------------------------------
// Problem constants (from reference): D=768, FF=3072, H=12, B=2, S=1025, M=8
// ---------------------------------------------------------------------------
typedef __attribute__((ext_vector_type(16))) __bf16          v16bf;
typedef __attribute__((ext_vector_type(8)))  float           v8f;
typedef __attribute__((ext_vector_type(8)))  unsigned short  u16x8;

union BF16Frag {
    v16bf v;
    u16x8 h[2];
    unsigned short u[16];
};

// Full fragment set for one 32-K step of a 32x64 wave tile.
struct FragSet { BF16Frag a0, a1, b0, b1, b2, b3; };

__device__ __forceinline__ unsigned short f2bf(float f) {
    union { float f; unsigned u; } cv; cv.f = f;
    unsigned r = cv.u + 0x7FFFu + ((cv.u >> 16) & 1u);   // round-to-nearest-even
    return (unsigned short)(r >> 16);
}

__device__ __forceinline__ void load_set(FragSet& f,
        const unsigned short* ap0, const unsigned short* ap1,
        const unsigned short* br0, const unsigned short* br1,
        const unsigned short* br2, const unsigned short* br3,
        int kk, int ofsA0, int ofsA1, int ofsB)
{
    f.a0.h[0] = *(const u16x8*)(ap0 + kk + ofsA0);
    f.a0.h[1] = *(const u16x8*)(ap0 + kk + ofsA1);
    f.a1.h[0] = *(const u16x8*)(ap1 + kk + ofsA0);
    f.a1.h[1] = *(const u16x8*)(ap1 + kk + ofsA1);
    f.b0.h[0] = *(const u16x8*)(br0 + kk + ofsB);
    f.b0.h[1] = *(const u16x8*)(br0 + kk + ofsB + 16);
    f.b1.h[0] = *(const u16x8*)(br1 + kk + ofsB);
    f.b1.h[1] = *(const u16x8*)(br1 + kk + ofsB + 16);
    f.b2.h[0] = *(const u16x8*)(br2 + kk + ofsB);
    f.b2.h[1] = *(const u16x8*)(br2 + kk + ofsB + 16);
    f.b3.h[0] = *(const u16x8*)(br3 + kk + ofsB);
    f.b3.h[1] = *(const u16x8*)(br3 + kk + ofsB + 16);
}

__device__ __forceinline__ void mma_set(v8f (&acc)[2][4], const FragSet& f)
{
    acc[0][0] = __builtin_amdgcn_wmma_f32_16x16x32_bf16(false, f.a0.v, false, f.b0.v, (short)0, acc[0][0], false, false);
    acc[1][0] = __builtin_amdgcn_wmma_f32_16x16x32_bf16(false, f.a1.v, false, f.b0.v, (short)0, acc[1][0], false, false);
    acc[0][1] = __builtin_amdgcn_wmma_f32_16x16x32_bf16(false, f.a0.v, false, f.b1.v, (short)0, acc[0][1], false, false);
    acc[1][1] = __builtin_amdgcn_wmma_f32_16x16x32_bf16(false, f.a1.v, false, f.b1.v, (short)0, acc[1][1], false, false);
    acc[0][2] = __builtin_amdgcn_wmma_f32_16x16x32_bf16(false, f.a0.v, false, f.b2.v, (short)0, acc[0][2], false, false);
    acc[1][2] = __builtin_amdgcn_wmma_f32_16x16x32_bf16(false, f.a1.v, false, f.b2.v, (short)0, acc[1][2], false, false);
    acc[0][3] = __builtin_amdgcn_wmma_f32_16x16x32_bf16(false, f.a0.v, false, f.b3.v, (short)0, acc[0][3], false, false);
    acc[1][3] = __builtin_amdgcn_wmma_f32_16x16x32_bf16(false, f.a1.v, false, f.b3.v, (short)0, acc[1][3], false, false);
}

// ---------------------------------------------------------------------------
// Batched WMMA GEMM:  C = act(alpha * A@B^T + bias) + residual
//   A  : Mr x K bf16 row-major (lda), K-contiguous
//   Bt : N  x K bf16 row-major (ldb), K-contiguous (pre-transposed)
//   C  : f32 (optional) and/or bf16 (optional), ldc
// Each wave computes a 32x64 strip (MT=2 x NT=4) = 8 v_wmma per 32-K step.
// Register double-buffering: the full fragment set for step k+32 is loaded
// BEFORE the 8 WMMAs of step k issue, so the matrix pipe overlaps VMEM and
// the waits only cover the previous prefetch. K must be a multiple of 64
// (caller pads). Out-of-range rows/cols are CLAMPED, not zeroed: a clamped
// A-row/B-col only feeds the same out-of-range C entry, which is never
// stored -> zero inner-loop predication, unconditional b128 loads.
// Fragment loads follow the ISA VGPR layouts:
//   A: lane(row) holds K = hi*8+0..7 and 16+hi*8+0..7  (two b128 loads)
//   B: lane(col) holds K = hi*8+0..7 and 16+hi*8+0..7  (two b128 loads)
// blockIdx.z decomposed as (outer=z/batchInner, inner=z%batchInner).
// ---------------------------------------------------------------------------
__global__ __launch_bounds__(256)
void wmma_gemm_bf16(const unsigned short* __restrict__ A,
                    const unsigned short* __restrict__ Bt,
                    const float* __restrict__ bias,
                    const float* __restrict__ Res,
                    float* __restrict__ C,
                    unsigned short* __restrict__ Cbf,
                    int Mr, int N, int K,
                    int lda, int ldb, int ldc,
                    int act, float alpha,
                    int batchInner,
                    long long sAo, long long sAi,
                    long long sBo, long long sBi,
                    long long sCo, long long sCi,
                    long long sRo, long long sRi)
{
    const int z  = blockIdx.z;
    const int bo = z / batchInner;
    const int bi = z % batchInner;
    A  += bo * sAo + bi * sAi;
    Bt += bo * sBo + bi * sBi;
    const long long coff = bo * sCo + bi * sCi;
    if (C)   C   += coff;
    if (Cbf) Cbf += coff;
    if (Res) Res += bo * sRo + bi * sRi;

    const int wave  = threadIdx.x >> 5;
    const int lane  = threadIdx.x & 31;
    const int hi    = lane >> 4;       // which 16-lane half
    const int idx16 = lane & 15;

    const int n0base = blockIdx.x * 64;
    const int m0 = (blockIdx.y * 8 + wave) * 32;   // 2 M-tiles per wave
    if (m0 >= Mr) return;              // wave-uniform exit (EXEC stays full)

    // Clamped row/col indices: unconditional loads, guarded stores.
    int ar0 = m0 + idx16;        if (ar0 >= Mr) ar0 = Mr - 1;
    int ar1 = m0 + 16 + idx16;   if (ar1 >= Mr) ar1 = Mr - 1;
    const unsigned short* ap0 = A + (long long)ar0 * lda;
    const unsigned short* ap1 = A + (long long)ar1 * lda;

    int bc0 = n0base + idx16;       if (bc0 >= N) bc0 = N - 1;
    int bc1 = n0base + 16 + idx16;  if (bc1 >= N) bc1 = N - 1;
    int bc2 = n0base + 32 + idx16;  if (bc2 >= N) bc2 = N - 1;
    int bc3 = n0base + 48 + idx16;  if (bc3 >= N) bc3 = N - 1;
    const unsigned short* br0 = Bt + (long long)bc0 * ldb;
    const unsigned short* br1 = Bt + (long long)bc1 * ldb;
    const unsigned short* br2 = Bt + (long long)bc2 * ldb;
    const unsigned short* br3 = Bt + (long long)bc3 * ldb;

    v8f acc[2][4] = {{{}, {}, {}, {}}, {{}, {}, {}, {}}};

    const int ofsA0 = hi * 8;          // first 8-element run of A fragment
    const int ofsA1 = 16 + hi * 8;     // second 8-element run of A fragment
    const int ofsB  = hi * 8;          // B frag runs at hi*8 and 16+hi*8

    // Double-buffered K loop; K is a multiple of 64.
    FragSet f0, f1;
    load_set(f0, ap0, ap1, br0, br1, br2, br3, 0, ofsA0, ofsA1, ofsB);
    for (int kk = 0; kk < K; kk += 64) {
        load_set(f1, ap0, ap1, br0, br1, br2, br3, kk + 32, ofsA0, ofsA1, ofsB);
        mma_set(acc, f0);
        if (kk + 64 < K)
            load_set(f0, ap0, ap1, br0, br1, br2, br3, kk + 64, ofsA0, ofsA1, ofsB);
        mma_set(acc, f1);
    }

#pragma unroll
    for (int mt = 0; mt < 2; ++mt) {
#pragma unroll
        for (int nt = 0; nt < 4; ++nt) {
            const int ng = n0base + nt * 16 + idx16;
            const bool nok = ng < N;
            const float cb = (nok && bias) ? bias[ng] : 0.f;
#pragma unroll
            for (int v = 0; v < 8; ++v) {
                // C/D layout: lanes 0-15 -> M=v, lanes 16-31 -> M=v+8
                const int mg = m0 + mt * 16 + v + hi * 8;
                if (mg < Mr && nok) {
                    float x = acc[mt][nt][v] * alpha + cb;
                    if (act == 1)  // exact GELU
                        x = 0.5f * x * (1.0f + erff(x * 0.70710678118654752f));
                    if (Res) x += Res[(long long)mg * ldc + ng];
                    if (C)   C[(long long)mg * ldc + ng] = x;
                    if (Cbf) Cbf[(long long)mg * ldc + ng] = f2bf(x);
                }
            }
        }
    }
}

// ---------------------------------------------------------------------------
// Row LayerNorm (eps=1e-6), optional f32 and bf16 outputs. 1 block per row.
// ---------------------------------------------------------------------------
__global__ __launch_bounds__(256)
void layernorm_kernel(const float* __restrict__ in,
                      const float* __restrict__ g,
                      const float* __restrict__ b,
                      float* __restrict__ outf,
                      unsigned short* __restrict__ outbf,
                      int rows, int d)
{
    __shared__ float red[256];
    const int row = blockIdx.x;
    if (row >= rows) return;
    const float* xr = in + (long long)row * d;

    float s = 0.f;
    for (int i = threadIdx.x; i < d; i += 256) s += xr[i];
    red[threadIdx.x] = s; __syncthreads();
    for (int o = 128; o > 0; o >>= 1) {
        if (threadIdx.x < o) red[threadIdx.x] += red[threadIdx.x + o];
        __syncthreads();
    }
    const float mean = red[0] / d; __syncthreads();

    float vv = 0.f;
    for (int i = threadIdx.x; i < d; i += 256) {
        const float t = xr[i] - mean; vv += t * t;
    }
    red[threadIdx.x] = vv; __syncthreads();
    for (int o = 128; o > 0; o >>= 1) {
        if (threadIdx.x < o) red[threadIdx.x] += red[threadIdx.x + o];
        __syncthreads();
    }
    const float rstd = rsqrtf(red[0] / d + 1e-6f);

    for (int i = threadIdx.x; i < d; i += 256) {
        const float y = (xr[i] - mean) * rstd * g[i] + b[i];
        if (outf)  outf[(long long)row * d + i] = y;
        if (outbf) outbf[(long long)row * d + i] = f2bf(y);
    }
}

// ---------------------------------------------------------------------------
// Row softmax of scores (optional -10000*(1-mask) bias; CLS key position 0
// masked in masked mode) -> bf16 probs, rows padded with zeros to Spad so
// the attn@V GEMM K-loop needs no tail predication. grid = (S, H, B).
// ---------------------------------------------------------------------------
__global__ __launch_bounds__(256)
void softmax_rows_kernel(const float* __restrict__ scores,
                         const int* __restrict__ masks,   // null => no mask
                         unsigned short* __restrict__ attn_bf,
                         int mIdx, int S, int Spad, int HWn, int Hn, int Mn)
{
    __shared__ float red[256];
    const int q = blockIdx.x, h = blockIdx.y, b = blockIdx.z;
    const float* row = scores + (((long long)b * Hn + h) * S + q) * (long long)S;
    const int* mrow = masks ? masks + ((long long)b * Mn + mIdx) * HWn : nullptr;

    float mx = -3.4e38f;
    for (int k = threadIdx.x; k < S; k += 256) {
        float s = row[k];
        if (mrow) s += (k == 0) ? -10000.f : (mrow[k - 1] ? 0.f : -10000.f);
        mx = fmaxf(mx, s);
    }
    red[threadIdx.x] = mx; __syncthreads();
    for (int o = 128; o > 0; o >>= 1) {
        if (threadIdx.x < o) red[threadIdx.x] = fmaxf(red[threadIdx.x], red[threadIdx.x + o]);
        __syncthreads();
    }
    mx = red[0]; __syncthreads();

    float sum = 0.f;
    for (int k = threadIdx.x; k < S; k += 256) {
        float s = row[k];
        if (mrow) s += (k == 0) ? -10000.f : (mrow[k - 1] ? 0.f : -10000.f);
        sum += __expf(s - mx);
    }
    red[threadIdx.x] = sum; __syncthreads();
    for (int o = 128; o > 0; o >>= 1) {
        if (threadIdx.x < o) red[threadIdx.x] += red[threadIdx.x + o];
        __syncthreads();
    }
    const float inv = 1.0f / red[0];

    unsigned short* orow = attn_bf + (((long long)b * Hn + h) * S + q) * (long long)Spad;
    for (int k = threadIdx.x; k < Spad; k += 256) {
        float p = 0.f;
        if (k < S) {
            float s = row[k];
            if (mrow) s += (k == 0) ? -10000.f : (mrow[k - 1] ? 0.f : -10000.f);
            p = __expf(s - mx) * inv;
        }
        orow[k] = f2bf(p);
    }
}

// f32 [K][N] -> bf16 transposed [N][K]  (weights become K-contiguous B^T)
__global__ void convert_transpose_kernel(const float* __restrict__ in,
                                         unsigned short* __restrict__ out,
                                         int K, int N)
{
    long long i = (long long)blockIdx.x * 256 + threadIdx.x;
    const long long total = (long long)K * N;
    const long long stride = (long long)gridDim.x * 256;
    for (; i < total; i += stride) {
        const int n = (int)(i / K), k = (int)(i % K);
        out[i] = f2bf(in[(long long)k * N + n]);
    }
}

// bf16 V [B][S][D] -> V^T [B][D][Spad], zero-padded rows s in [S, Spad)
__global__ void transpose_pad_v_kernel(const unsigned short* __restrict__ in,
                                       unsigned short* __restrict__ out,
                                       int Bn, int S, int D, int Spad)
{
    long long i = (long long)blockIdx.x * 256 + threadIdx.x;
    const long long total = (long long)Bn * D * Spad;
    const long long stride = (long long)gridDim.x * 256;
    for (; i < total; i += stride) {
        const int s = (int)(i % Spad);
        const long long bd = i / Spad;
        const int d = (int)(bd % D);
        const int b = (int)(bd / D);
        out[i] = (s < S) ? in[((long long)b * S + s) * D + d] : (unsigned short)0;
    }
}

__global__ void write_idx_kernel(float* out, int n, int m)
{
    const int i = threadIdx.x;
    if (i < n) out[i] = (float)(i / m);
}

// ---------------------------------------------------------------------------
// Host orchestration
// ---------------------------------------------------------------------------
extern "C" void kernel_launch(void* const* d_in, const int* in_sizes, int n_in,
                              void* d_out, int out_size, void* d_ws, size_t ws_size,
                              hipStream_t stream)
{
    constexpr int Bn = 2, Sn = 1025, Dn = 768, Hn = 12, HDn = 64, FFn = 3072;
    constexpr int Mk = 8, HWn = 1024;
    constexpr int Spad = 1088;            // S padded to multiple of 64
    constexpr int BS  = Bn * Sn;          // 2050
    constexpr int BMS = Bn * Mk * Sn;     // 16400
    const long long SS2 = (long long)Sn * Sn;
    const long long SSp = (long long)Sn * Spad;
    const long long XO  = (long long)BS * Dn;
    const long long XN  = (long long)BMS * Dn;

    const float* x     = (const float*)d_in[0];
    const int*   masks = (const int*)  d_in[1];
    const float* Wq = (const float*)d_in[2];  const float* bq = (const float*)d_in[3];
    const float* Wk = (const float*)d_in[4];  const float* bk = (const float*)d_in[5];
    const float* Wv = (const float*)d_in[6];  const float* bv = (const float*)d_in[7];
    const float* Wp = (const float*)d_in[8];  const float* bp = (const float*)d_in[9];
    const float* W1 = (const float*)d_in[10]; const float* b1 = (const float*)d_in[11];
    const float* W2 = (const float*)d_in[12]; const float* b2 = (const float*)d_in[13];
    const float* g1 = (const float*)d_in[14]; const float* be1 = (const float*)d_in[15];
    const float* g2 = (const float*)d_in[16]; const float* be2 = (const float*)d_in[17];

    float* xout = (float*)d_out;          // [BS, D]
    float* xnew = (float*)d_out + XO;     // [BMS, D]

    char* wp_ = (char*)d_ws;
    auto alloc = [&](size_t bytes) -> void* {
        void* p = (void*)wp_;
        wp_ += (bytes + 255) & ~(size_t)255;
        return p;
    };
    unsigned short* wqt  = (unsigned short*)alloc((size_t)Dn * Dn * 2);
    unsigned short* wkt  = (unsigned short*)alloc((size_t)Dn * Dn * 2);
    unsigned short* wvt  = (unsigned short*)alloc((size_t)Dn * Dn * 2);
    unsigned short* wpt  = (unsigned short*)alloc((size_t)Dn * Dn * 2);
    unsigned short* w1t  = (unsigned short*)alloc((size_t)Dn * FFn * 2);   // [FF][D]
    unsigned short* w2t  = (unsigned short*)alloc((size_t)FFn * Dn * 2);   // [D][FF]
    float*          xn   = (float*)         alloc((size_t)BS * Dn * 4);
    unsigned short* xn_bf= (unsigned short*)alloc((size_t)BS * Dn * 2);
    unsigned short* q_bf = (unsigned short*)alloc((size_t)BS * Dn * 2);
    unsigned short* k_bf = (unsigned short*)alloc((size_t)BS * Dn * 2);
    unsigned short* v_bf = (unsigned short*)alloc((size_t)BS * Dn * 2);
    unsigned short* vt_bf= (unsigned short*)alloc((size_t)Bn * Dn * Spad * 2);
    float*          scores=(float*)         alloc((size_t)Bn * Hn * SS2 * 4);
    unsigned short* at_bf= (unsigned short*)alloc((size_t)Bn * Hn * SSp * 2);
    unsigned short* h_bf = (unsigned short*)alloc((size_t)BS * Dn * 2);
    unsigned short* ln2_bf=(unsigned short*)alloc((size_t)BS * Dn * 2);
    unsigned short* t1_bf= (unsigned short*)alloc((size_t)BS * FFn * 2);
    unsigned short* nh_bf= (unsigned short*)alloc((size_t)BMS * Dn * 2);
    unsigned short* ln2n_bf=(unsigned short*)alloc((size_t)BMS * Dn * 2);
    unsigned short* t1n_bf= (unsigned short*)alloc((size_t)BMS * FFn * 2);
    (void)ws_size; (void)n_in; (void)in_sizes; (void)out_size;

    auto gemm = [&](const unsigned short* A, const unsigned short* Bt,
                    const float* bias, const float* Res,
                    float* C, unsigned short* Cbf,
                    int Mr, int N, int K, int lda, int ldb, int ldc,
                    int act, float alpha,
                    int batch, int batchInner,
                    long long sAo, long long sAi, long long sBo, long long sBi,
                    long long sCo, long long sCi, long long sRo, long long sRi) {
        dim3 grid((N + 63) / 64, (Mr + 255) / 256, batch);
        wmma_gemm_bf16<<<grid, 256, 0, stream>>>(A, Bt, bias, Res, C, Cbf,
            Mr, N, K, lda, ldb, ldc, act, alpha,
            batchInner, sAo, sAi, sBo, sBi, sCo, sCi, sRo, sRi);
    };

    // 1) weights -> bf16, transposed to [N][K] so B fragments are K-contiguous
    convert_transpose_kernel<<<512, 256, 0, stream>>>(Wq, wqt, Dn, Dn);
    convert_transpose_kernel<<<512, 256, 0, stream>>>(Wk, wkt, Dn, Dn);
    convert_transpose_kernel<<<512, 256, 0, stream>>>(Wv, wvt, Dn, Dn);
    convert_transpose_kernel<<<512, 256, 0, stream>>>(Wp, wpt, Dn, Dn);
    convert_transpose_kernel<<<512, 256, 0, stream>>>(W1, w1t, Dn, FFn);
    convert_transpose_kernel<<<512, 256, 0, stream>>>(W2, w2t, FFn, Dn);

    // 2) LN1 -> xn (f32, residual broadcast) + xn_bf (GEMM operand)
    layernorm_kernel<<<BS, 256, 0, stream>>>(x, g1, be1, xn, xn_bf, BS, Dn);

    // 3) Q, K, V projections (bf16 outputs)
    gemm(xn_bf, wqt, bq, nullptr, nullptr, q_bf, BS, Dn, Dn, Dn, Dn, Dn,
         0, 1.f, 1, 1, 0,0, 0,0, 0,0, 0,0);
    gemm(xn_bf, wkt, bk, nullptr, nullptr, k_bf, BS, Dn, Dn, Dn, Dn, Dn,
         0, 1.f, 1, 1, 0,0, 0,0, 0,0, 0,0);
    gemm(xn_bf, wvt, bv, nullptr, nullptr, v_bf, BS, Dn, Dn, Dn, Dn, Dn,
         0, 1.f, 1, 1, 0,0, 0,0, 0,0, 0,0);

    // 3b) V -> V^T [B][D][Spad] (zero-padded) for the attn@V GEMMs
    transpose_pad_v_kernel<<<1024, 256, 0, stream>>>(v_bf, vt_bf, Bn, Sn, Dn, Spad);

    // 4) scores = Q @ K^T / 8, batched over (b,h); K matrix already N x K
    gemm(q_bf, k_bf, nullptr, nullptr, scores, nullptr,
         Sn, Sn, HDn, Dn, Dn, Sn, 0, 0.125f,
         Bn * Hn, Hn,
         (long long)Sn * Dn, 64, (long long)Sn * Dn, 64,
         (long long)Hn * SS2, SS2, 0, 0);

    // 5) main softmax -> bf16 probs (rows padded to Spad)
    softmax_rows_kernel<<<dim3(Sn, Hn, Bn), 256, 0, stream>>>(
        scores, nullptr, at_bf, 0, Sn, Spad, HWn, Hn, Mk);

    // 6) h = attn @ V (per head): A lda=Spad, B = V^T head rows, ldb=Spad
    gemm(at_bf, vt_bf, nullptr, nullptr, nullptr, h_bf,
         Sn, HDn, Spad, Spad, Spad, Dn, 0, 1.f,
         Bn * Hn, Hn,
         (long long)Hn * SSp, SSp, (long long)Dn * Spad, (long long)64 * Spad,
         (long long)Sn * Dn, 64, 0, 0);

    // 7) x_out = x + h @ Wp + bp
    gemm(h_bf, wpt, bp, x, xout, nullptr, BS, Dn, Dn, Dn, Dn, Dn,
         0, 1.f, 1, 1, 0,0, 0,0, 0,0, 0,0);

    // 8-10) main MLP with residual
    layernorm_kernel<<<BS, 256, 0, stream>>>(xout, g2, be2, nullptr, ln2_bf, BS, Dn);
    gemm(ln2_bf, w1t, b1, nullptr, nullptr, t1_bf, BS, FFn, Dn, Dn, Dn, FFn,
         /*gelu*/1, 1.f, 1, 1, 0,0, 0,0, 0,0, 0,0);
    gemm(t1_bf, w2t, b2, xout, xout, nullptr, BS, Dn, FFn, FFn, FFn, Dn,
         0, 1.f, 1, 1, 0,0, 0,0, 0,0, 0,0);

    // 11) masked branch: reuse scores; per-mask softmax + attn@V
    for (int m = 0; m < Mk; ++m) {
        softmax_rows_kernel<<<dim3(Sn, Hn, Bn), 256, 0, stream>>>(
            scores, masks, at_bf, m, Sn, Spad, HWn, Hn, Mk);
        gemm(at_bf, vt_bf, nullptr, nullptr, nullptr,
             nh_bf + (long long)m * Sn * Dn,
             Sn, HDn, Spad, Spad, Spad, Dn, 0, 1.f,
             Bn * Hn, Hn,
             (long long)Hn * SSp, SSp, (long long)Dn * Spad, (long long)64 * Spad,
             (long long)Mk * Sn * Dn, 64, 0, 0);
    }

    // 12) x_new = xn(broadcast over m) + new_h @ Wp + bp   (batch = B*M)
    gemm(nh_bf, wpt, bp, xn, xnew, nullptr,
         Sn, Dn, Dn, Dn, Dn, Dn, 0, 1.f,
         Bn * Mk, Mk,
         (long long)Mk * Sn * Dn, (long long)Sn * Dn, 0, 0,
         (long long)Mk * Sn * Dn, (long long)Sn * Dn,
         (long long)Sn * Dn, 0);

    // 13-15) masked MLP with residual
    layernorm_kernel<<<BMS, 256, 0, stream>>>(xnew, g2, be2, nullptr, ln2n_bf, BMS, Dn);
    gemm(ln2n_bf, w1t, b1, nullptr, nullptr, t1n_bf, BMS, FFn, Dn, Dn, Dn, FFn,
         /*gelu*/1, 1.f, 1, 1, 0,0, 0,0, 0,0, 0,0);
    gemm(t1n_bf, w2t, b2, xnew, xnew, nullptr, BMS, Dn, FFn, FFn, FFn, Dn,
         0, 1.f, 1, 1, 0,0, 0,0, 0,0, 0,0);

    // 16) img_idx = repeat(arange(B), M)
    write_idx_kernel<<<1, 32, 0, stream>>>((float*)d_out + XO + XN, Bn * Mk, Mk);
}
